// GCNII_2310692405651
// MI455X (gfx1250) — matrix operand order repaired and checked by reading, not compiled, not run on hardware
//
#include <hip/hip_runtime.h>
#include <math.h>

// ---------------------------------------------------------------------------
// GCNII forward for MI455X (gfx1250): bf16 WMMA GEMMs + L2-resident SpMM.
// ---------------------------------------------------------------------------

#define F_IN 512
#define HIDC 128
#define NCLS 64

typedef __attribute__((ext_vector_type(16))) __bf16 v16bf;
typedef __attribute__((ext_vector_type(8)))  float  v8f;

// ---------------------------------------------------------------------------
// Degree / normalization kernels
// ---------------------------------------------------------------------------
__global__ void deg_init_kernel(float* __restrict__ deg, int n) {
    int i = blockIdx.x * blockDim.x + threadIdx.x;
    if (i < n) deg[i] = 1.0f;                     // +1 for self loop
}

__global__ void deg_accum_kernel(const int* __restrict__ dst,
                                 float* __restrict__ deg, int E) {
    int e = blockIdx.x * blockDim.x + threadIdx.x;
    if (e < E) unsafeAtomicAdd(&deg[dst[e]], 1.0f);
}

__global__ void dinv_selfw_kernel(const float* __restrict__ deg,
                                  float* __restrict__ dinv,
                                  float* __restrict__ selfw, int n) {
    int i = blockIdx.x * blockDim.x + threadIdx.x;
    if (i < n) {
        float d = rsqrtf(deg[i]);
        dinv[i]  = d;
        selfw[i] = d * d;
    }
}

__global__ void edgew_kernel(const int* __restrict__ src,
                             const int* __restrict__ dst,
                             const float* __restrict__ dinv,
                             float* __restrict__ ew, int E) {
    int e = blockIdx.x * blockDim.x + threadIdx.x;
    if (e < E) ew[e] = dinv[src[e]] * dinv[dst[e]];
}

// p[n,:] = selfw[n] * h[n,:]   (initializes the SpMM accumulator)
__global__ void self_scale_kernel(const float* __restrict__ h,
                                  const float* __restrict__ selfw,
                                  float* __restrict__ p, long total) {
    long i = (long)blockIdx.x * blockDim.x + threadIdx.x;
    if (i < total) p[i] = h[i] * selfw[i >> 7];   // HID = 128
}

// One wave per edge: 32 lanes x float4 = 128 features, float scatter-atomics.
// h/p/ew/indices are L2-resident (192 MB), so this is L2-atomic bound.
__global__ __launch_bounds__(256) void spmm_edges_kernel(
    const int* __restrict__ src, const int* __restrict__ dst,
    const float* __restrict__ ew, const float* __restrict__ h,
    float* __restrict__ p, int E)
{
    long i = (long)blockIdx.x * blockDim.x + threadIdx.x;
    int e = (int)(i >> 5);
    int lane = (int)(i & 31);
    if (e >= E) return;
    float w = ew[e];
    int s = src[e], d = dst[e];
    const float4* hs = reinterpret_cast<const float4*>(h + (long)s * HIDC);
    float4 v = hs[lane];
    float* pd = p + (long)d * HIDC + lane * 4;
    unsafeAtomicAdd(pd + 0, v.x * w);
    unsafeAtomicAdd(pd + 1, v.y * w);
    unsafeAtomicAdd(pd + 2, v.z * w);
    unsafeAtomicAdd(pd + 3, v.w * w);
}

// ---------------------------------------------------------------------------
// WMMA GEMM, 128-row x 128-col tile per 256-thread (8-wave) block.
// MODE 0:  Out = relu(A0 @ W + bias)                       (input layer, K=512)
// MODE 1:  t = (1-alpha)*A0 + alpha*X0  (A0 = p)
//          Out = relu((1-beta)*t + beta*(t @ W))           (GCNII layer, K=128)
// A/B converted f32->bf16 for V_WMMA_F32_16X16X32_BF16; accumulate in f32.
// All 8 B fragments are preloaded (16x ds_load_b128) before the WMMA burst so
// the 8 v_wmma ops issue back-to-back instead of stalling on dscnt each.
// ---------------------------------------------------------------------------
template<int K, int MODE>
__global__ __launch_bounds__(256) void gemm_wmma_kernel(
    const float* __restrict__ A0,   // [Nrows x K]   (x, or p)
    const float* __restrict__ X0,   // [Nrows x 128] (x0; MODE1 only)
    const float* __restrict__ W,    // [K x 128] row-major
    const float* __restrict__ bias, // [128]         (MODE0 only)
    float* __restrict__ Out,        // [Nrows x 128]
    int Nrows, float alpha, float beta)
{
    __shared__ __bf16 Wt[HIDC * 128];   // W^T slice: [n][k_local], 32 KB
    __shared__ __bf16 At[128 * 32];     // A tile:    [row][k_local], 8 KB

    const int tid  = threadIdx.x;
    const int wave = tid >> 5;
    const int lane = tid & 31;
    const int row0 = blockIdx.x * 128;

    v8f acc[8];
    for (int j = 0; j < 8; ++j)
        for (int r = 0; r < 8; ++r) acc[j][r] = 0.0f;

    for (int kc = 0; kc < K; kc += 128) {
        __syncthreads();  // previous compute done before overwriting Wt
        // Stage 128x128 weight slice transposed into LDS as bf16
        for (int i = tid; i < 128 * HIDC; i += 256) {
            int kk = i >> 7;        // local k
            int n  = i & 127;
            Wt[n * 128 + kk] = (__bf16)W[(long)(kc + kk) * HIDC + n];
        }
        for (int kb = 0; kb < 128; kb += 32) {
            __syncthreads();  // Wt visible / previous At consumers done
            // Stage 128x32 activation tile into LDS as bf16
            for (int i = tid; i < 128 * 32; i += 256) {
                int r  = i >> 5;
                int k  = i & 31;
                int gm = row0 + r;
                float v = 0.0f;
                if (gm < Nrows) {
                    if (MODE == 0) {
                        v = A0[(long)gm * K + (kc + kb + k)];
                    } else {
                        long off = (long)gm * HIDC + (kc + kb + k);
                        v = (1.0f - alpha) * A0[off] + alpha * X0[off];
                    }
                }
                At[i] = (__bf16)v;
            }
            __syncthreads();

            // A fragment (16x32 bf16, ISA 7.12.2 layout): lane half selects
            // K {0..7,16..23} vs {8..15,24..31}; M = lane&15.
            const int mrow = wave * 16 + (lane & 15);
            const int k0   = (lane < 16) ? 0 : 8;
            v16bf afrag;
            const __bf16* ap = &At[mrow * 32 + k0];
            #pragma unroll
            for (int i = 0; i < 8; ++i) {
                afrag[i]     = ap[i];
                afrag[i + 8] = ap[16 + i];
            }

            // Preload ALL 8 B fragments (independent of the WMMAs), then fire
            // the 8 WMMAs back-to-back.  B layout: lane half selects K
            // {0..7,16..23} vs {8..15,24..31}; N = lane&15; from W^T in LDS.
            const int kk0 = (lane < 16) ? 0 : 8;
            v16bf bfr[8];
            #pragma unroll
            for (int j = 0; j < 8; ++j) {
                const __bf16* bp = &Wt[(j * 16 + (lane & 15)) * 128 + kb + kk0];
                #pragma unroll
                for (int i = 0; i < 8; ++i) { bfr[j][i] = bp[i]; bfr[j][i + 8] = bp[16 + i]; }
            }
            #pragma unroll
            for (int j = 0; j < 8; ++j) {
                acc[j] = __builtin_amdgcn_wmma_f32_16x16x32_bf16(
                    false, afrag, false, bfr[j], (short)0, acc[j], false, false);
            }
        }
    }

    // Epilogue: C layout — VGPR r holds M=r (lanes 0-15) / M=r+8 (16-31),
    // N = lane&15 within each 16-wide tile.
    const int nl    = lane & 15;
    const int mbase = wave * 16 + ((lane < 16) ? 0 : 8);
    for (int j = 0; j < 8; ++j) {
        const int n = j * 16 + nl;
        #pragma unroll
        for (int r = 0; r < 8; ++r) {
            const int gm = row0 + mbase + r;
            if (gm >= Nrows) continue;
            float v = acc[j][r];
            float o;
            if (MODE == 0) {
                o = v + bias[n];
            } else {
                long off = (long)gm * HIDC + n;
                float t  = (1.0f - alpha) * A0[off] + alpha * X0[off];
                o = (1.0f - beta) * t + beta * v;
            }
            Out[(long)gm * HIDC + n] = fmaxf(o, 0.0f);
        }
    }
}

// ---------------------------------------------------------------------------
// Head: logits = z @ W_out + b_out ; log_softmax over 64 classes.
// One 64-thread block per node; LDS tree reductions for max / sum.
// ---------------------------------------------------------------------------
__global__ __launch_bounds__(64) void head_kernel(
    const float* __restrict__ z, const float* __restrict__ Wout,
    const float* __restrict__ bout, float* __restrict__ out, int Nrows)
{
    const int node = blockIdx.x;
    const int j = threadIdx.x;
    __shared__ float zs[HIDC];
    __shared__ float red[NCLS];

    zs[j]      = z[(long)node * HIDC + j];
    zs[j + 64] = z[(long)node * HIDC + 64 + j];
    __syncthreads();

    float a = bout[j];
    #pragma unroll 8
    for (int k = 0; k < HIDC; ++k) a = fmaf(zs[k], Wout[k * NCLS + j], a);

    red[j] = a;
    __syncthreads();
    for (int off = 32; off >= 1; off >>= 1) {
        if (j < off) red[j] = fmaxf(red[j], red[j + off]);
        __syncthreads();
    }
    float m = red[0];
    __syncthreads();
    red[j] = expf(a - m);
    __syncthreads();
    for (int off = 32; off >= 1; off >>= 1) {
        if (j < off) red[j] += red[j + off];
        __syncthreads();
    }
    float lse = m + logf(red[0]);
    out[(long)node * NCLS + j] = a - lse;
}

// ---------------------------------------------------------------------------
// Host orchestration
// ---------------------------------------------------------------------------
extern "C" void kernel_launch(void* const* d_in, const int* in_sizes, int n_in,
                              void* d_out, int out_size, void* d_ws, size_t ws_size,
                              hipStream_t stream)
{
    const float* x       = (const float*)d_in[0];
    const int*   eidx    = (const int*)  d_in[1];
    const float* W_in    = (const float*)d_in[2];
    const float* b_in    = (const float*)d_in[3];
    const float* W_convs = (const float*)d_in[4];
    const float* W_out   = (const float*)d_in[5];
    const float* b_out   = (const float*)d_in[6];

    const int N = in_sizes[0] / F_IN;
    const int E = in_sizes[1] / 2;
    const int L = in_sizes[4] / (HIDC * HIDC);
    const float ALPHA = 0.1f;
    const float THETA = 0.5f;

    const int* src = eidx;
    const int* dst = eidx + E;

    // Workspace: deg, selfw, dinv, ew, x0, p  (~110 MB)
    float* ws    = (float*)d_ws;
    float* deg   = ws;  ws += N;
    float* selfw = ws;  ws += N;
    float* dinv  = ws;  ws += N;
    float* ew    = ws;  ws += E;
    float* x0    = ws;  ws += (size_t)N * HIDC;
    float* p     = ws;  ws += (size_t)N * HIDC;

    // h lives in the z-region of d_out; log-softmax follows it.
    float* h     = (float*)d_out;
    float* outls = (float*)d_out + (size_t)N * HIDC;

    const int nb  = (N + 255) / 256;
    const int ebk = (E + 255) / 256;

    // Normalization: deg = 1 + indegree; dinv = deg^-1/2; ew = dinv_s*dinv_d
    deg_init_kernel  <<<nb,  256, 0, stream>>>(deg, N);
    deg_accum_kernel <<<ebk, 256, 0, stream>>>(dst, deg, E);
    dinv_selfw_kernel<<<nb,  256, 0, stream>>>(deg, dinv, selfw, N);
    edgew_kernel     <<<ebk, 256, 0, stream>>>(src, dst, dinv, ew, E);

    const int gblocks = (N + 127) / 128;

    // x0 = relu(x @ W_in + b_in)  — WMMA, K = 512
    gemm_wmma_kernel<F_IN, 0><<<gblocks, 256, 0, stream>>>(
        x, nullptr, W_in, b_in, x0, N, 0.0f, 0.0f);

    const long total  = (long)N * HIDC;
    const int  sblk   = (int)((total + 255) / 256);
    const int  spblk  = (int)(((long)E * 32 + 255) / 256);

    for (int l = 0; l < L; ++l) {
        const float beta = logf(THETA / (float)(l + 1) + 1.0f);
        const float* hcur = (l == 0) ? x0 : h;

        // p = selfw .* h ; then scatter-add edge contributions
        self_scale_kernel<<<sblk,  256, 0, stream>>>(hcur, selfw, p, total);
        spmm_edges_kernel<<<spblk, 256, 0, stream>>>(src, dst, ew, hcur, p, E);

        // h = relu((1-beta)*t + beta*(t @ W_l)),  t = (1-a)*p + a*x0  (fused)
        gemm_wmma_kernel<HIDC, 1><<<gblocks, 256, 0, stream>>>(
            p, x0, W_convs + (size_t)l * HIDC * HIDC, nullptr, h, N, ALPHA, beta);
    }

    // logits + log_softmax
    head_kernel<<<N, 64, 0, stream>>>(h, W_out, b_out, outls, N);
}
